// ScaledDotProductAttention_50663434224570
// MI455X (gfx1250) — matrix operand order repaired and checked by reading, not compiled, not run on hardware
//
#include <hip/hip_runtime.h>

typedef __attribute__((ext_vector_type(16))) _Float16 v16h;
typedef __attribute__((ext_vector_type(8)))  float    v8f;

#define LSEQ 1024
#define DH   64
#define LSTR 1028   // logits row stride (f32), 16B pad
#define VRSTR 66    // raw V chunk row stride (f32): 66%64=2 -> conflict-free col gather
#define SCALE 0.125f  // 1/sqrt(64)

// A-matrix fragment (16x32 f16). Layout matches B: lane holds 16 CONSECUTIVE
// K values: lanes 0-15 (hi=0) K=0..15, lanes 16-31 (hi=1) K=16..31.
__device__ __forceinline__ v16h afrag_from_f32(const float* p, int hi) {
  const float* p0 = p + 16 * hi;
  v16h a;
#pragma unroll
  for (int i = 0; i < 16; ++i) {
    a[i]     = (_Float16)p0[i];
  }
  return a;
}

// B-matrix fragment (32x16 f16). Layout used here (same order as A):
// lane holds 16 CONSECUTIVE K values: lanes 0-15 K=0..15, lanes 16-31 K=16..31.
// p points at B-column data laid out contiguously in K (a row of K, or of V^T).
__device__ __forceinline__ v16h bfrag_from_f32(const float* p, int hi) {
  const float* p0 = p + 16 * hi;
  v16h a;
#pragma unroll
  for (int i = 0; i < 16; ++i) a[i] = (_Float16)p0[i];
  return a;
}

// Async-copy one 32x64 f32 V chunk (8KB, contiguous in global) into an LDS
// buffer with row stride VRSTR, via global_load_async_to_lds_b64 (ASYNCcnt).
__device__ __forceinline__ void async_copy_v_chunk(const float* gsrc,
                                                   const float* lbuf, int t) {
  const unsigned lbase = (unsigned)(uintptr_t)lbuf;   // low 32 bits = LDS offset
  const unsigned long long gbase = (unsigned long long)(uintptr_t)gsrc;
#pragma unroll
  for (int i = 0; i < 8; ++i) {
    const int e    = t + 128 * i;   // f32-pair index; consecutive threads coalesce
    const int krow = e >> 5;        // 32 pairs per 64-f32 row
    const int cp   = e & 31;
    const unsigned loff = lbase + (unsigned)((krow * VRSTR + 2 * cp) * 4);
    const unsigned long long g = gbase + (unsigned long long)e * 8ull;
    asm volatile("global_load_async_to_lds_b64 %0, %1, off"
                 :: "v"(loff), "v"(g) : "memory");
  }
}

__global__ __launch_bounds__(128)
void sdpa_fwd_kernel(const float* __restrict__ q, const float* __restrict__ kmat,
                     const float* __restrict__ vmat,
                     const unsigned char* __restrict__ mask,
                     float* __restrict__ out, float* __restrict__ attn) {
  __shared__ float s_log[16 * LSTR];       // 16 x 1024 logits / probs (f32)
  __shared__ float s_vraw[2][32 * VRSTR];  // double-buffered raw f32 V chunks
  __shared__ float s_red[16 * 8];
  __shared__ float s_rowmax[16];
  __shared__ float s_rowinv[16];

  const int b     = blockIdx.y;
  const int qbase = blockIdx.x * 16;
  const int t     = threadIdx.x;
  const int lane  = t & 31;
  const int w     = t >> 5;        // wave id 0..3
  const int m     = lane & 15;
  const int hi    = lane >> 4;

  const float* qtile = q    + ((size_t)b * LSEQ + qbase) * DH;
  const float* kbat  = kmat + (size_t)b * LSEQ * DH;
  const float* vbat  = vmat + (size_t)b * LSEQ * DH;

  // ---- Phase 0: Q A-fragments (16x64 tile -> two 16x32 k-chunks) ----
  const v16h aq0 = afrag_from_f32(qtile + m * DH + 0,  hi);
  const v16h aq1 = afrag_from_f32(qtile + m * DH + 32, hi);

  // ---- Phase 1: S = (Q K^T) * scale, waves split the 64 key tiles ----
  // B-fragment needs columns of K^T = rows of K: no transpose required.
  for (int nt = w; nt < 64; nt += 4) {
    const float* krow = kbat + (size_t)(nt * 16 + m) * DH;
    __builtin_prefetch(kbat + (size_t)((((nt + 4) & 63) * 16) + m) * DH, 0, 1);
    const v16h bk0 = bfrag_from_f32(krow + 0,  hi);
    const v16h bk1 = bfrag_from_f32(krow + 32, hi);
    v8f c = {};
    c = __builtin_amdgcn_wmma_f32_16x16x32_f16(false, aq0, false, bk0,
                                               (short)0, c, false, false);
    c = __builtin_amdgcn_wmma_f32_16x16x32_f16(false, aq1, false, bk1,
                                               (short)0, c, false, false);
    const int col = nt * 16 + m;
#pragma unroll
    for (int r = 0; r < 8; ++r)
      s_log[(r + 8 * hi) * LSTR + col] = c[r] * SCALE;
  }
  __syncthreads();

  // ---- Phase 2: apply mask (coalesced global reads), row max ----
  const int row = t >> 3;   // 16 rows x 8 threads
  const int j   = t & 7;
  float* lrow = s_log + row * LSTR;
  const unsigned char* mrow = mask + ((size_t)b * LSEQ + qbase + row) * LSEQ;
  const float NEGINF = -__builtin_inff();
  float rmax = NEGINF;
#pragma unroll
  for (int blk = 0; blk < 8; ++blk) {
    const int cb = blk * 128 + j * 16;
    const uint4 mu = *(const uint4*)(mrow + cb);
    float fs[16];
#pragma unroll
    for (int i = 0; i < 4; ++i)
      *(float4*)(fs + 4 * i) = *(const float4*)(lrow + cb + 4 * i);
    const unsigned int mw[4] = {mu.x, mu.y, mu.z, mu.w};
#pragma unroll
    for (int i = 0; i < 16; ++i) {
      if ((mw[i >> 2] >> ((i & 3) * 8)) & 0xFFu) fs[i] = NEGINF;
      rmax = fmaxf(rmax, fs[i]);
    }
#pragma unroll
    for (int i = 0; i < 4; ++i)
      *(float4*)(lrow + cb + 4 * i) = *(const float4*)(fs + 4 * i);
  }
  s_red[row * 8 + j] = rmax;
  __syncthreads();
  if (t < 16) {
    float mx = s_red[t * 8];
#pragma unroll
    for (int i = 1; i < 8; ++i) mx = fmaxf(mx, s_red[t * 8 + i]);
    s_rowmax[t] = mx;
  }
  __syncthreads();

  // ---- Phase 3: unnormalized exp + row sum ----
  const float rm = s_rowmax[row];
  float rsum = 0.0f;
#pragma unroll
  for (int blk = 0; blk < 8; ++blk) {
    const int cb = blk * 128 + j * 16;
#pragma unroll
    for (int i = 0; i < 16; ++i) {
      const float p = __expf(lrow[cb + i] - rm);
      lrow[cb + i] = p;
      rsum += p;
    }
  }
  s_red[row * 8 + j] = rsum;
  __syncthreads();
  if (t < 16) {
    float s = s_red[t * 8];
#pragma unroll
    for (int i = 1; i < 8; ++i) s += s_red[t * 8 + i];
    s_rowinv[t] = 1.0f / s;
  }
  __syncthreads();

  // ---- Phase 4: write normalized attn to HBM (dominant traffic) ----
  float* arow = attn + ((size_t)b * LSEQ + qbase + row) * LSEQ;
  const float rinv = s_rowinv[row];
#pragma unroll
  for (int blk = 0; blk < 8; ++blk) {
    const int cb = blk * 128 + j * 16;
#pragma unroll
    for (int i = 0; i < 4; ++i) {
      float4 pv = *(const float4*)(lrow + cb + 4 * i);
      pv.x *= rinv; pv.y *= rinv; pv.z *= rinv; pv.w *= rinv;
      *(float4*)(arow + cb + 4 * i) = pv;
    }
  }

  // ---- Phase 5: O = P_unnorm @ V, async double-buffered 32-row V chunks ----
  v8f oc = {};
  const float* pmrow = s_log + m * LSTR;
  const int dcol = w * 16 + m;            // this wave's D column for B frags
  async_copy_v_chunk(vbat, s_vraw[0], t); // prefetch chunk 0
  for (int kc = 0; kc < 32; ++kc) {
    asm volatile("s_wait_asynccnt 0x0" ::: "memory");  // my chunk-kc ops landed
    __syncthreads();  // everyone's landed; prior-iter reads of other buf done
    if (kc + 1 < 32)
      async_copy_v_chunk(vbat + (size_t)(kc + 1) * 32 * DH,
                         s_vraw[(kc + 1) & 1], t);
    const float* vb = s_vraw[kc & 1];
    const v16h ap = afrag_from_f32(pmrow + kc * 32, hi);
    // B fragment: 16 consecutive K rows of this V chunk at column dcol;
    // stride-66 rows -> 16 distinct LDS banks, conflict-free.
    v16h bv;
#pragma unroll
    for (int i = 0; i < 16; ++i)
      bv[i] = (_Float16)vb[(16 * hi + i) * VRSTR + dcol];
    oc = __builtin_amdgcn_wmma_f32_16x16x32_f16(false, ap, false, bv,
                                                (short)0, oc, false, false);
  }

  // ---- Phase 6: normalize rows of O and store ----
  float* obase = out + ((size_t)b * LSEQ + qbase) * DH;
#pragma unroll
  for (int r = 0; r < 8; ++r) {
    const int mr = r + 8 * hi;
    obase[(size_t)mr * DH + dcol] = oc[r] * s_rowinv[mr];
  }
}

extern "C" void kernel_launch(void* const* d_in, const int* in_sizes, int n_in,
                              void* d_out, int out_size, void* d_ws, size_t ws_size,
                              hipStream_t stream) {
  const float* q = (const float*)d_in[0];
  const float* k = (const float*)d_in[1];
  const float* v = (const float*)d_in[2];
  const unsigned char* mask = (const unsigned char*)d_in[3];

  const int B = in_sizes[0] / (LSEQ * DH);   // 64 for the reference shapes
  float* out  = (float*)d_out;
  float* attn = out + (size_t)B * LSEQ * DH; // tuple: (output, attn) concatenated

  dim3 grid(LSEQ / 16, B);
  sdpa_fwd_kernel<<<grid, 128, 0, stream>>>(q, k, v, mask, out, attn);
}
